// AttentionLePE_3951369912845
// MI455X (gfx1250) — matrix operand order repaired and checked by reading, not compiled, not run on hardware
//
#include <hip/hip_runtime.h>
#include <cstdint>
#include <cstddef>

// ---------------------------------------------------------------------------
// Problem constants (reference: B=4, H=W=56, C=256, NUM_HEADS=8)
// ---------------------------------------------------------------------------
#define B_BATCH 4
#define HDIM    56
#define WDIM    56
#define CDIM    256
#define NPIX    (HDIM * WDIM)     // 3136
#define SEQ     (NPIX + 1)        // 3137 (cls + pixels)
#define SEQP    3200              // SEQ padded to multiple of 64
#define NHEAD   8
#define HD      32                // head dim
#define QKV_N   (3 * CDIM)        // 768
#define QKW     512               // bf16 qk row width (q | k)
#define MTOT    (B_BATCH * SEQ)   // 12548 rows total
#define MPAD    (MTOT + 64)       // padded rows for safe 64-key overreads

typedef __attribute__((ext_vector_type(8)))  float        v8f;
typedef __attribute__((ext_vector_type(16))) __bf16       v16bf;
typedef __attribute__((ext_vector_type(8)))  __bf16       v8bf;
typedef __attribute__((ext_vector_type(4)))  unsigned int u32x4;

// fp32 -> bf16 round-to-nearest-even (raw u16 bits)
static __device__ __forceinline__ unsigned short f2b(float f) {
  unsigned int u = __builtin_bit_cast(unsigned int, f);
  u += 0x7FFFu + ((u >> 16) & 1u);
  return (unsigned short)(u >> 16);
}

// Build a 16-element bf16 fragment from two 16-byte runs (LDS or global).
static __device__ __forceinline__ v16bf load_frag(const unsigned short* p0,
                                                  const unsigned short* p1) {
  v8bf lo = __builtin_bit_cast(v8bf, *reinterpret_cast<const u32x4*>(p0));
  v8bf hi = __builtin_bit_cast(v8bf, *reinterpret_cast<const u32x4*>(p1));
  return __builtin_shufflevector(lo, hi, 0, 1, 2, 3, 4, 5, 6, 7,
                                          8, 9, 10, 11, 12, 13, 14, 15);
}

// ---------------------------------------------------------------------------
// GEMM:  out[m][n] = sum_k A[m][k] * W[n][k]  (+ bias, mode 1)
//   mode 0: A = concat(cls,x), W = qkv_w (768x256).
//           n <  512 : bf16 -> qk[m][n]          (q | k, row-major)
//           n >= 512 : bf16 -> vT[b][head][d][s] (V transposed)
//   mode 1: A = yin (attn+lepe), W = proj_w, +bias, scatter fp32 to d_out.
// Block: 128 threads (4 waves). Block tile 16(M) x 64(N). K stepped by 32.
// ---------------------------------------------------------------------------
__global__ __launch_bounds__(128)
void gemm16_bf16_wmma(const float* __restrict__ x,
                      const float* __restrict__ cls,
                      const float* __restrict__ yin,
                      const float* __restrict__ w,
                      const float* __restrict__ bias,
                      unsigned short* __restrict__ qk_out,
                      unsigned short* __restrict__ vt_out,
                      float* __restrict__ out_x,
                      float* __restrict__ out_cls,
                      int mode) {
  __shared__ __align__(16) unsigned short Ab[16][32];
  __shared__ __align__(16) unsigned short Bb[64][32];

  const int tid  = threadIdx.x;
  const int wv   = tid >> 5;
  const int lane = tid & 31;
  const int hf   = lane >> 4;
  const int l16  = lane & 15;
  const int m0   = blockIdx.x * 16;
  const int nblk = blockIdx.y * 64;

  v8f acc = {};

  // A staging: 128 threads cover 16 rows x 32 cols (4 cols each).
  const int arow = tid >> 3;
  const int acol = (tid & 7) * 4;
  const float* asrc = nullptr;
  {
    int gm = m0 + arow;
    if (gm < MTOT) {
      if (mode == 0) {
        int b = gm / SEQ, s = gm % SEQ;
        asrc = (s == 0) ? (cls + (size_t)b * CDIM)
                        : (x + ((size_t)b * NPIX + (s - 1)) * CDIM);
      } else {
        asrc = yin + (size_t)gm * CDIM;
      }
    }
  }
  // B staging: 64 rows(n) x 32 cols, each thread does 16 cols.
  const int bn  = tid >> 1;
  const int bcb = (tid & 1) * 16;
  const float* bsrc = w + (size_t)(nblk + bn) * CDIM + bcb;

  for (int k0 = 0; k0 < CDIM; k0 += 32) {
    if (asrc) {
      float4 f = *reinterpret_cast<const float4*>(asrc + k0 + acol);
      Ab[arow][acol + 0] = f2b(f.x);
      Ab[arow][acol + 1] = f2b(f.y);
      Ab[arow][acol + 2] = f2b(f.z);
      Ab[arow][acol + 3] = f2b(f.w);
    } else {
      Ab[arow][acol + 0] = 0; Ab[arow][acol + 1] = 0;
      Ab[arow][acol + 2] = 0; Ab[arow][acol + 3] = 0;
    }
#pragma unroll
    for (int j = 0; j < 16; j += 4) {
      float4 f = *reinterpret_cast<const float4*>(bsrc + k0 + j);
      Bb[bn][bcb + j + 0] = f2b(f.x);
      Bb[bn][bcb + j + 1] = f2b(f.y);
      Bb[bn][bcb + j + 2] = f2b(f.z);
      Bb[bn][bcb + j + 3] = f2b(f.w);
    }
    __syncthreads();

    v16bf a = load_frag(&Ab[l16][8 * hf], &Ab[l16][8 * hf + 16]);
    int brow = wv * 16 + l16;
    v16bf bm = load_frag(&Bb[brow][8 * hf], &Bb[brow][8 * hf + 16]);

    acc = __builtin_amdgcn_wmma_f32_16x16x32_bf16(false, a, false, bm,
                                                  (short)0, acc, false, false);
    __syncthreads();
  }

  // Epilogue. D layout: element r -> (m = r + 8*hf, n = l16).
  const int gn = nblk + wv * 16 + l16;
#pragma unroll
  for (int r = 0; r < 8; ++r) {
    int gm = m0 + r + 8 * hf;
    if (gm >= MTOT) continue;
    float v = acc[r];
    if (mode == 0) {
      unsigned short hv = f2b(v);
      if (gn < QKW) {                       // q | k region (wave-uniform test)
        qk_out[(size_t)gm * QKW + gn] = hv;
      } else {                              // V -> transposed per (b, head)
        int b = gm / SEQ, s = gm % SEQ;
        int c = gn - QKW;
        vt_out[(((size_t)b * NHEAD + (c >> 5)) * HD + (c & 31)) * SEQP + s] = hv;
      }
    } else {
      v += bias[gn];
      int b = gm / SEQ, s = gm % SEQ;
      if (s == 0) out_cls[(size_t)b * CDIM + gn] = v;
      else        out_x[((size_t)b * NPIX + (s - 1)) * CDIM + gn] = v;
    }
  }
}

// ---------------------------------------------------------------------------
// Flash attention: one wave per (16-query tile, head, batch).
// 64 keys per iteration: 4 QK WMMAs + 4 PV WMMAs share ONE online-softmax
// reduction/rescale, halving per-key shuffle/exp bookkeeping vs 32-key
// blocks.  Q/K/V fragments load directly from global bf16 (contiguous
// 16B runs per lane); only P round-trips through 2KB of LDS.
// Logits are pre-scaled by log2(e) so exp2f (native v_exp_f32) is exact.
// ---------------------------------------------------------------------------
__global__ __launch_bounds__(32)
void flash_attn_wmma(const unsigned short* __restrict__ qk,
                     const unsigned short* __restrict__ vt,
                     float* __restrict__ y) {
  __shared__ __align__(16) unsigned short Pb[16][64];

  const int lane = threadIdx.x;
  const int hf   = lane >> 4;
  const int l16  = lane & 15;
  const int q0   = blockIdx.x * 16;
  const int head = blockIdx.y;
  const int b    = blockIdx.z;
  const float scl2 = 0.17677669529663687f * 1.4426950408889634f; // hd^-.5 * log2(e)

  // Q A-fragment straight from global (padded buffer tolerates overread;
  // garbage rows stay row-isolated and are never written back).
  const unsigned short* qp =
      qk + ((size_t)b * SEQ + q0 + l16) * QKW + head * HD + 16 * hf;
  const v16bf aq = load_frag(qp, qp + 8);

  // Per-lane fragment base pointers: K rows l16 / +16 / +32 / +48 of each
  // 64-key block; V^T dim columns l16 and l16+16.
  const unsigned short* kp0 =
      qk + ((size_t)b * SEQ + l16) * QKW + CDIM + head * HD + 16 * hf;
  const unsigned short* vr0 =
      vt + (((size_t)b * NHEAD + head) * HD + l16) * SEQP + 16 * hf;
  const unsigned short* vr1 = vr0 + (size_t)16 * SEQP;

  v8f acc0 = {}, acc1 = {};
  float mrow[8], lrow[8];
#pragma unroll
  for (int r = 0; r < 8; ++r) { mrow[r] = -1.0e30f; lrow[r] = 0.0f; }

  for (int kv0 = 0; kv0 < SEQ; kv0 += 64) {
    const unsigned short* ka = kp0 + (size_t)kv0 * QKW;
    const unsigned short* va = vr0 + kv0;
    const unsigned short* vb = vr1 + kv0;

    if (kv0 + 64 < SEQ) {  // hide next-block latency (global_prefetch_b8)
      __builtin_prefetch(ka + (size_t)64 * QKW, 0, 1);
      __builtin_prefetch(va + 64, 0, 1);
    }

    // Scores: S = Q (16x32) x K^T, four 16-key column tiles.
    v16bf bk0 = load_frag(ka,                       ka + 8);
    v16bf bk1 = load_frag(ka + (size_t)16 * QKW,    ka + (size_t)16 * QKW + 8);
    v16bf bk2 = load_frag(ka + (size_t)32 * QKW,    ka + (size_t)32 * QKW + 8);
    v16bf bk3 = load_frag(ka + (size_t)48 * QKW,    ka + (size_t)48 * QKW + 8);
    v8f z = {};
    v8f s0 = __builtin_amdgcn_wmma_f32_16x16x32_bf16(false, aq, false, bk0,
                                                     (short)0, z, false, false);
    v8f s1 = __builtin_amdgcn_wmma_f32_16x16x32_bf16(false, aq, false, bk1,
                                                     (short)0, z, false, false);
    v8f s2 = __builtin_amdgcn_wmma_f32_16x16x32_bf16(false, aq, false, bk2,
                                                     (short)0, z, false, false);
    v8f s3 = __builtin_amdgcn_wmma_f32_16x16x32_bf16(false, aq, false, bk3,
                                                     (short)0, z, false, false);

    const bool ok0 = (kv0 + l16) < SEQ;
    const bool ok1 = (kv0 + 16 + l16) < SEQ;
    const bool ok2 = (kv0 + 32 + l16) < SEQ;
    const bool ok3 = (kv0 + 48 + l16) < SEQ;
#pragma unroll
    for (int r = 0; r < 8; ++r) {
      float a0 = ok0 ? s0[r] * scl2 : -1.0e30f;
      float a1 = ok1 ? s1[r] * scl2 : -1.0e30f;
      float a2 = ok2 ? s2[r] * scl2 : -1.0e30f;
      float a3 = ok3 ? s3[r] * scl2 : -1.0e30f;
      // Row m reductions stay inside one 16-lane half (xor masks < 16).
      float t = fmaxf(fmaxf(a0, a1), fmaxf(a2, a3));
      t = fmaxf(t, __shfl_xor(t, 1, 32));
      t = fmaxf(t, __shfl_xor(t, 2, 32));
      t = fmaxf(t, __shfl_xor(t, 4, 32));
      t = fmaxf(t, __shfl_xor(t, 8, 32));
      float nm = fmaxf(mrow[r], t);
      float p0 = exp2f(a0 - nm);
      float p1 = exp2f(a1 - nm);
      float p2 = exp2f(a2 - nm);
      float p3 = exp2f(a3 - nm);
      float rs = (p0 + p1) + (p2 + p3);
      rs += __shfl_xor(rs, 1, 32);
      rs += __shfl_xor(rs, 2, 32);
      rs += __shfl_xor(rs, 4, 32);
      rs += __shfl_xor(rs, 8, 32);
      float corr = exp2f(mrow[r] - nm);
      lrow[r] = lrow[r] * corr + rs;
      mrow[r] = nm;
      acc0[r] *= corr;
      acc1[r] *= corr;
      // Spill P (D layout) to LDS for re-read in A layout.
      int m = r + 8 * hf;
      Pb[m][l16]      = f2b(p0);
      Pb[m][l16 + 16] = f2b(p1);
      Pb[m][l16 + 32] = f2b(p2);
      Pb[m][l16 + 48] = f2b(p3);
    }
    __syncthreads();

    // O += P (16x64) x V (64x32): two P A-fragments x two V^T column tiles.
    v16bf ap0 = load_frag(&Pb[l16][16 * hf],      &Pb[l16][16 * hf + 8]);
    v16bf ap1 = load_frag(&Pb[l16][32 + 16 * hf], &Pb[l16][32 + 16 * hf + 8]);
    v16bf bv0 = load_frag(va,      va + 8);
    v16bf bv1 = load_frag(vb,      vb + 8);
    v16bf bv2 = load_frag(va + 32, va + 40);
    v16bf bv3 = load_frag(vb + 32, vb + 40);
    acc0 = __builtin_amdgcn_wmma_f32_16x16x32_bf16(false, ap0, false, bv0,
                                                   (short)0, acc0, false, false);
    acc0 = __builtin_amdgcn_wmma_f32_16x16x32_bf16(false, ap1, false, bv2,
                                                   (short)0, acc0, false, false);
    acc1 = __builtin_amdgcn_wmma_f32_16x16x32_bf16(false, ap0, false, bv1,
                                                   (short)0, acc1, false, false);
    acc1 = __builtin_amdgcn_wmma_f32_16x16x32_bf16(false, ap1, false, bv3,
                                                   (short)0, acc1, false, false);
    __syncthreads();
  }

  // Normalize, write head slice of attention output (fp32 for LePE add).
#pragma unroll
  for (int r = 0; r < 8; ++r) {
    int s = q0 + r + 8 * hf;
    if (s >= SEQ) continue;
    float inv = 1.0f / lrow[r];
    float* dst = y + ((size_t)b * SEQ + s) * CDIM + head * HD;
    dst[l16]      = acc0[r] * inv;
    dst[16 + l16] = acc1[r] * inv;
  }
}

// ---------------------------------------------------------------------------
// LePE: depthwise 5x5 conv (SAME) + bias, accumulated into y rows s >= 1.
// ---------------------------------------------------------------------------
__global__ __launch_bounds__(256)
void lepe_add(const float* __restrict__ x, const float* __restrict__ lw,
              const float* __restrict__ lb, float* __restrict__ y) {
  size_t idx = (size_t)blockIdx.x * 256 + threadIdx.x;
  if (idx >= (size_t)B_BATCH * NPIX * CDIM) return;
  int c = (int)(idx & (CDIM - 1));
  size_t p = idx >> 8;
  int wq = (int)(p % WDIM);
  size_t p2 = p / WDIM;
  int hq = (int)(p2 % HDIM);
  int b  = (int)(p2 / HDIM);

  float sum = lb[c];
#pragma unroll
  for (int i = 0; i < 5; ++i) {
    int hh = hq + i - 2;
    if (hh < 0 || hh >= HDIM) continue;
#pragma unroll
    for (int j = 0; j < 5; ++j) {
      int ww = wq + j - 2;
      if (ww < 0 || ww >= WDIM) continue;
      sum += x[(((size_t)b * HDIM + hh) * WDIM + ww) * CDIM + c] *
             lw[c * 25 + i * 5 + j];
    }
  }
  y[((size_t)b * SEQ + 1 + (size_t)hq * WDIM + wq) * CDIM + c] += sum;
}

// ---------------------------------------------------------------------------
// Launch
// ---------------------------------------------------------------------------
extern "C" void kernel_launch(void* const* d_in, const int* in_sizes, int n_in,
                              void* d_out, int out_size, void* d_ws,
                              size_t ws_size, hipStream_t stream) {
  (void)in_sizes; (void)n_in; (void)out_size; (void)ws_size;
  const float* x      = (const float*)d_in[0];  // (4,56,56,256)
  const float* cls    = (const float*)d_in[1];  // (4,1,256)
  const float* qkv_w  = (const float*)d_in[2];  // (768,256)
  const float* proj_w = (const float*)d_in[3];  // (256,256)
  const float* proj_b = (const float*)d_in[4];  // (256,)
  const float* lepe_w = (const float*)d_in[5];  // (256,1,5,5)
  const float* lepe_b = (const float*)d_in[6];  // (256,)

  float* out_x   = (float*)d_out;                           // B*N*C
  float* out_cls = out_x + (size_t)B_BATCH * NPIX * CDIM;   // B*C

  // Workspace layout (16B-aligned sections):
  //   qk : MPAD x 512 bf16 (q | k)          12.9 MB
  //   vT : B*NHEAD*HD x SEQP bf16            6.6 MB
  //   y  : MTOT x 256 fp32                  12.8 MB
  unsigned short* qk = (unsigned short*)d_ws;
  unsigned short* vt = qk + (size_t)MPAD * QKW;
  float*          y  = (float*)(vt + (size_t)B_BATCH * NHEAD * HD * SEQP);

  const int mtiles = (MTOT + 15) / 16;  // 785

  // 1) QKV projection -> bf16 qk rows + transposed bf16 V
  gemm16_bf16_wmma<<<dim3(mtiles, QKV_N / 64), 128, 0, stream>>>(
      x, cls, nullptr, qkv_w, nullptr, qk, vt, nullptr, nullptr, 0);

  // 2) Flash attention -> y (fp32)
  flash_attn_wmma<<<dim3((SEQ + 15) / 16, NHEAD, B_BATCH), 32, 0, stream>>>(
      qk, vt, y);

  // 3) LePE depthwise conv added into y (pixel rows only)
  lepe_add<<<dim3((B_BATCH * NPIX * CDIM) / 256), 256, 0, stream>>>(
      x, lepe_w, lepe_b, y);

  // 4) Output projection + bias, scatter to (x_out, cls_out)
  gemm16_bf16_wmma<<<dim3(mtiles, CDIM / 64), 128, 0, stream>>>(
      nullptr, nullptr, y, proj_w, proj_b, nullptr, nullptr, out_x, out_cls, 1);
}
